// RPN_84207128805815
// MI455X (gfx1250) — compile-verified, matrix operand order checked by reading, not run on hardware
//
#include <hip/hip_runtime.h>
#include <math.h>

typedef __attribute__((ext_vector_type(16))) _Float16 v16h;
typedef __attribute__((ext_vector_type(8)))  float    v8f;

#define NTOT_CAT 4720
#define NSORT    8192
#define W148     148

__device__ const int g_kper[5] = {1000, 1000, 1000, 1000, 720};

// ---- CDNA5 async global->LDS helpers (ASYNCcnt-tracked) ----
__device__ __forceinline__ void async_copy_b128(unsigned lds_off, const void* gaddr) {
    asm volatile("global_load_async_to_lds_b128 %0, %1, off"
                 :: "v"(lds_off), "v"(gaddr)
                 : "memory");
}
__device__ __forceinline__ void wait_async0() {
    asm volatile("s_wait_asynccnt 0" ::: "memory");
}
__device__ __forceinline__ unsigned lds_off_of(const void* p) {
    // flat LDS address: upper 32 bits = shared aperture, low 32 bits = LDS byte offset
    return (unsigned)(uintptr_t)p;
}

// ---------------- layout conversion kernels ----------------

// NCHW fp32 -> NHWC f16 (per level)
__global__ void k_convert(const float* __restrict__ src, _Float16* __restrict__ dst, int hw) {
    int tid = blockIdx.x * blockDim.x + threadIdx.x;
    int n = hw * 256;
    if (tid >= n) return;
    int p = tid >> 8;
    int c = tid & 255;
    dst[tid] = (_Float16)src[c * hw + p];
}

// conv_w [O=256][I=256][3][3] fp32 -> wpackT [O=256][K=2304] f16 where K = tap*256 + I
__global__ void k_packw(const float* __restrict__ cw, _Float16* __restrict__ wp) {
    int tid = blockIdx.x * blockDim.x + threadIdx.x;
    if (tid >= 256 * 2304) return;
    int o = tid / 2304;
    int r = tid % 2304;
    int tap = r >> 8;
    int i = r & 255;
    wp[tid] = (_Float16)cw[o * 2304 + i * 9 + tap];
}

// ---------------- 3x3 conv as implicit GEMM via WMMA ----------------
// A: [M = spatial][K = 2304] gathered on the fly (im2col), B: wpackT [N][K].
// Block: 256 thr = 8 waves; tile 128(M) x 128(N); k-step 32; double-buffered LDS.
// Each wave: 2 M-subs x 4 N-subs = 8 WMMAs per k-step (A and B fragments reused).
// B tile staged with global_load_async_to_lds_b128 (never OOB); A tile via
// registers (border zero-fill), with its global loads issued before the WMMAs.
__global__ __launch_bounds__(256) void k_conv(const _Float16* __restrict__ xh,
                                              const _Float16* __restrict__ wp,
                                              const float* __restrict__ bias,
                                              _Float16* __restrict__ feat,
                                              int H, int Wd, int hw) {
    __shared__ _Float16 As[2][128 * 40];  // [m][k], stride 40 halves (80B)
    __shared__ _Float16 Bs[2][128 * 40];  // [n][k], stride 40 halves

    const int tid  = threadIdx.x;
    const int lane = tid & 31;
    const int wv   = tid >> 5;
    const int m0   = blockIdx.x * 128;
    const int n0   = blockIdx.y * 128;

    v8f acc[2][4] = {};

    // A-tile load coords: thread -> (row m, 16-half chunk)
    const int am   = tid >> 1;           // 0..127
    const int ach  = (tid & 1) * 16;     // 0 or 16 (halves)
    const int apos = m0 + am;
    const int ay   = apos / Wd;
    const int ax   = apos % Wd;

    // B-tile load coords: thread -> (col n, 16-half half-row)
    const int bn  = tid >> 1;            // 0..127
    const int bko = (tid & 1) * 16;      // 0 or 16 (halves)
    const _Float16* wrow = wp + (size_t)(n0 + bn) * 2304 + bko;

    // fragment coords: wave -> 2 M-subs, 4 N-subs
    const int msb = (wv >> 1) * 2;       // 0,2,4,6
    const int ng  = wv & 1;              // 0..1
    const int kb  = (lane < 16) ? 0 : 8;
    const int mrow0 = (msb + 0) * 16 + (lane & 15);
    const int mrow1 = (msb + 1) * 16 + (lane & 15);

    // per-thread LDS staging addresses (byte offsets) for both buffers
    unsigned boffL[2];
    boffL[0] = lds_off_of(&Bs[0][bn * 40 + bko]);
    boffL[1] = lds_off_of(&Bs[1][bn * 40 + bko]);

    // ---- helpers ----
    auto loadA = [&](int b, uint4* av) {
        const int tap = b >> 3;
        const int c0  = (b & 7) * 32;
        const int ky  = tap / 3 - 1;
        const int kx  = tap % 3 - 1;
        av[0] = make_uint4(0u, 0u, 0u, 0u);
        av[1] = make_uint4(0u, 0u, 0u, 0u);
        const int ny = ay + ky, nx = ax + kx;
        if (apos < hw && ny >= 0 && ny < H && nx >= 0 && nx < Wd) {
            const uint4* src = (const uint4*)(xh + ((size_t)(ny * Wd + nx) * 256 + c0 + ach));
            av[0] = src[0];
            av[1] = src[1];
        }
    };
    auto storeA = [&](int buf, const uint4* av) {
        uint4* dst = (uint4*)(&As[buf][am * 40 + ach]);
        dst[0] = av[0];
        dst[1] = av[1];
    };
    auto stageB = [&](int b, int buf) {
        const _Float16* src = wrow + (size_t)b * 32;
        async_copy_b128(boffL[buf],      (const void*)src);
        async_copy_b128(boffL[buf] + 16, (const void*)(src + 8));
    };

    // ---- prologue: stage tile 0 into buffer 0 ----
    {
        uint4 a0[2];
        loadA(0, a0);
        stageB(0, 0);
        storeA(0, a0);
        wait_async0();
    }
    __syncthreads();

    for (int b = 0; b < 72; ++b) {
        const int cur = b & 1;
        const int nxt = cur ^ 1;
        const bool more = (b + 1) < 72;

        // issue next tile's global traffic before the math
        uint4 anext[2];
        if (more) {
            loadA(b + 1, anext);
            stageB(b + 1, nxt);
        }

        // ---- fragments + WMMA from current buffer ----
        v16h afrag[2];
        {
            const _Float16* ap0 = &As[cur][mrow0 * 40 + kb];
            ((uint4*)&afrag[0])[0] = *(const uint4*)ap0;
            ((uint4*)&afrag[0])[1] = *(const uint4*)(ap0 + 16);
            const _Float16* ap1 = &As[cur][mrow1 * 40 + kb];
            ((uint4*)&afrag[1])[0] = *(const uint4*)ap1;
            ((uint4*)&afrag[1])[1] = *(const uint4*)(ap1 + 16);
        }
        v16h bfrag[4];
#pragma unroll
        for (int j = 0; j < 4; ++j) {
            const int ncol = (ng * 4 + j) * 16 + (lane & 15);
            const _Float16* bp = &Bs[cur][ncol * 40 + kb];
            ((uint4*)&bfrag[j])[0] = *(const uint4*)bp;
            ((uint4*)&bfrag[j])[1] = *(const uint4*)(bp + 16);
        }
#pragma unroll
        for (int i = 0; i < 2; ++i) {
#pragma unroll
            for (int j = 0; j < 4; ++j) {
                acc[i][j] = __builtin_amdgcn_wmma_f32_16x16x32_f16(
                    false, afrag[i], false, bfrag[j], (short)0, acc[i][j], false, false);
            }
        }

        // commit A for next tile after the math
        if (more) {
            storeA(nxt, anext);
        }
        wait_async0();
        __syncthreads();
    }

    // ---- epilogue: bias + ReLU -> f16 NHWC feature ----
#pragma unroll
    for (int i = 0; i < 2; ++i) {
#pragma unroll
        for (int j = 0; j < 4; ++j) {
            const int n = n0 + (ng * 4 + j) * 16 + (lane & 15);
            const float bv = bias[n];
#pragma unroll
            for (int r = 0; r < 8; ++r) {
                const int ml  = (msb + i) * 16 + ((lane >> 4) * 8) + r;
                const int pos = m0 + ml;
                if (pos < hw) {
                    float v = acc[i][j][r] + bv;
                    v = v > 0.f ? v : 0.f;
                    feat[(size_t)pos * 256 + n] = (_Float16)v;
                }
            }
        }
    }
}

// ---------------- 1x1 heads + softmax + box decode ----------------
__global__ void k_heads(const _Float16* __restrict__ feat,
                        const float* __restrict__ ow, const float* __restrict__ ob,
                        const float* __restrict__ dw, const float* __restrict__ db,
                        float* __restrict__ scores, float* __restrict__ boxes,
                        int hw, int Wd, float stride, float asize, int n) {
    int j = blockIdx.x * blockDim.x + threadIdx.x;
    if (j >= n) return;
    int a = j / hw;
    int p = j % hw;
    const _Float16* f = feat + (size_t)p * 256;
    const float* w_l0 = ow + a * 256;
    const float* w_l1 = ow + (3 + a) * 256;
    const float* w_dx = dw + a * 256;
    const float* w_dy = dw + (3 + a) * 256;
    const float* w_dw = dw + (6 + a) * 256;
    const float* w_dh = dw + (9 + a) * 256;
    float l0 = ob[a], l1 = ob[3 + a];
    float dx = db[a], dy = db[3 + a], dwv = db[6 + a], dhv = db[9 + a];
    for (int c = 0; c < 256; ++c) {
        float fv = (float)f[c];
        l0  += fv * w_l0[c];
        l1  += fv * w_l1[c];
        dx  += fv * w_dx[c];
        dy  += fv * w_dy[c];
        dwv += fv * w_dw[c];
        dhv += fv * w_dh[c];
    }
    float sc = 1.f / (1.f + expf(l0 - l1));   // softmax row 1 of 2
    // anchor
    float ratio = (a == 0) ? 0.5f : (a == 1 ? 1.f : 2.f);
    float sq = sqrtf(ratio);
    float aw = asize / sq, ah = asize * sq;
    int y = p / Wd, x = p % Wd;
    float cx = (x + 0.5f) * stride, cy = (y + 0.5f) * stride;
    // delta -> box
    dwv = fminf(fmaxf(dwv, -4.135f), 4.135f);
    dhv = fminf(fmaxf(dhv, -4.135f), 4.135f);
    float pcx = dx * aw + cx, pcy = dy * ah + cy;
    float pw = expf(dwv) * aw, ph = expf(dhv) * ah;
    float x1 = pcx - 0.5f * pw, y1 = pcy - 0.5f * ph;
    float x2 = pcx + 0.5f * pw, y2 = pcy + 0.5f * ph;
    x1 = fminf(fmaxf(x1, 0.f), 1280.f);
    x2 = fminf(fmaxf(x2, 0.f), 1280.f);
    y1 = fminf(fmaxf(y1, 0.f), 768.f);
    y2 = fminf(fmaxf(y2, 0.f), 768.f);
    bool valid = (x2 > x1) && (y2 > y1);
    scores[j] = valid ? sc : -1.f;
    boxes[(size_t)j * 4 + 0] = x1;
    boxes[(size_t)j * 4 + 1] = y1;
    boxes[(size_t)j * 4 + 2] = x2;
    boxes[(size_t)j * 4 + 3] = y2;
}

// ---------------- per-level top-k: histogram select ----------------
__device__ __forceinline__ int score_bin(float s) {
    if (!(s > 0.f)) return 0;
    unsigned b = __float_as_uint(s);
    return (int)(b >> 19);   // < 2033, positive floats monotone in bits
}

__global__ void k_histzero(int* hist) {
    int i = blockIdx.x * blockDim.x + threadIdx.x;
    if (i < 5 * 4096) hist[i] = 0;
}

__global__ void k_hist(const float* __restrict__ sc, int* __restrict__ hist, int n) {
    int i = blockIdx.x * blockDim.x + threadIdx.x;
    if (i >= n) return;
    atomicAdd(&hist[score_bin(sc[i])], 1);
}

__global__ void k_thresh(const int* __restrict__ hist, int* selT, int* selR, int* selC) {
    int l = blockIdx.x;
    if (threadIdx.x != 0) return;
    const int* h = hist + l * 4096;
    int k = g_kper[l];
    int above = 0, t = 0;
    for (int b = 4095; b >= 0; --b) {
        int c = h[b];
        if (above + c >= k) { t = b; break; }
        above += c;
    }
    selT[l] = t;
    selR[l] = k - above;
    selC[2 * l] = 0;
    selC[2 * l + 1] = 0;
}

__global__ void k_compact(const float* __restrict__ sc, const float* __restrict__ bx,
                          float* __restrict__ selsc, float* __restrict__ selbox,
                          const int* __restrict__ selT, const int* __restrict__ selR,
                          int* __restrict__ selC,
                          int level, int n, int catoffk, int kk) {
    int j = blockIdx.x * blockDim.x + threadIdx.x;
    if (j >= n) return;
    float s = sc[j];
    int b = score_bin(s);
    int t = selT[level];
    int dst = -1;
    if (b > t) {
        dst = atomicAdd(&selC[2 * level], 1);
    } else if (b == t) {
        int s2 = atomicAdd(&selC[2 * level + 1], 1);
        int rem = selR[level];
        if (s2 < rem) dst = (kk - rem) + s2;
    }
    if (dst >= 0) {
        int o = catoffk + dst;
        selsc[o] = s;
        selbox[(size_t)o * 4 + 0] = bx[(size_t)j * 4 + 0];
        selbox[(size_t)o * 4 + 1] = bx[(size_t)j * 4 + 1];
        selbox[(size_t)o * 4 + 2] = bx[(size_t)j * 4 + 2];
        selbox[(size_t)o * 4 + 3] = bx[(size_t)j * 4 + 3];
    }
}

// ---------------- global descending sort (bitonic, GMEM) ----------------
__global__ void k_sortinit(float* keys, int* idx, const float* __restrict__ selsc) {
    int i = blockIdx.x * blockDim.x + threadIdx.x;
    if (i >= NSORT) return;
    keys[i] = (i < NTOT_CAT) ? selsc[i] : -3.0f;
    idx[i] = i;
}

__global__ void k_bitonic(float* key, int* idx, int j, int k2) {
    int i = blockIdx.x * blockDim.x + threadIdx.x;
    if (i >= NSORT) return;
    int p = i ^ j;
    if (p > i) {
        bool desc = ((i & k2) == 0);
        float a = key[i], b = key[p];
        bool sw = desc ? (a < b) : (a > b);
        if (sw) {
            key[i] = b; key[p] = a;
            int t = idx[i]; idx[i] = idx[p]; idx[p] = t;
        }
    }
}

__global__ void k_gather(const float* __restrict__ keys, const int* __restrict__ idx,
                         const float* __restrict__ selbox,
                         float* __restrict__ sbox, float* __restrict__ ssc) {
    int i = blockIdx.x * blockDim.x + threadIdx.x;
    if (i >= NTOT_CAT) return;
    ssc[i] = keys[i];
    int q = idx[i];
    if (q < NTOT_CAT) {
        sbox[(size_t)i * 4 + 0] = selbox[(size_t)q * 4 + 0];
        sbox[(size_t)i * 4 + 1] = selbox[(size_t)q * 4 + 1];
        sbox[(size_t)i * 4 + 2] = selbox[(size_t)q * 4 + 2];
        sbox[(size_t)i * 4 + 3] = selbox[(size_t)q * 4 + 3];
    } else {
        sbox[(size_t)i * 4 + 0] = 0.f;
        sbox[(size_t)i * 4 + 1] = 0.f;
        sbox[(size_t)i * 4 + 2] = 0.f;
        sbox[(size_t)i * 4 + 3] = 0.f;
    }
}

// ---------------- NMS: suppression bitmask + sequential scan ----------------
__global__ void k_supmask(const float* __restrict__ sbox, unsigned* __restrict__ sup) {
    int tid = blockIdx.x * blockDim.x + threadIdx.x;
    if (tid >= NTOT_CAT * W148) return;
    int i  = tid / W148;
    int wd = tid % W148;
    float ix1 = sbox[(size_t)i * 4 + 0], iy1 = sbox[(size_t)i * 4 + 1];
    float ix2 = sbox[(size_t)i * 4 + 2], iy2 = sbox[(size_t)i * 4 + 3];
    float ai = (ix2 - ix1) * (iy2 - iy1);
    unsigned bits = 0;
    for (int b = 0; b < 32; ++b) {
        int j = wd * 32 + b;
        if (j >= i) break;
        float jx1 = sbox[(size_t)j * 4 + 0], jy1 = sbox[(size_t)j * 4 + 1];
        float jx2 = sbox[(size_t)j * 4 + 2], jy2 = sbox[(size_t)j * 4 + 3];
        float aj = (jx2 - jx1) * (jy2 - jy1);
        float lx = fmaxf(ix1, jx1), ly = fmaxf(iy1, jy1);
        float rx = fminf(ix2, jx2), ry = fminf(iy2, jy2);
        float w = fmaxf(rx - lx, 0.f), h = fmaxf(ry - ly, 0.f);
        float inter = w * h;
        float iou = inter / (ai + aj - inter + 1e-9f);
        if (iou > 0.7f) bits |= (1u << b);
    }
    sup[(size_t)i * W148 + wd] = bits;
}

__global__ void k_nms(const unsigned* __restrict__ sup, unsigned* __restrict__ keepOut) {
    __shared__ unsigned keepw[W148];
    __shared__ int flag;
    for (int t = threadIdx.x; t < W148; t += blockDim.x) keepw[t] = 0;
    __syncthreads();
    for (int i = 0; i < NTOT_CAT; ++i) {
        if (threadIdx.x == 0) flag = 0;
        __syncthreads();
        unsigned part = 0;
        for (int w = threadIdx.x; w < W148; w += blockDim.x)
            part |= sup[(size_t)i * W148 + w] & keepw[w];
        if (part) atomicOr(&flag, 1);
        __syncthreads();
        if (threadIdx.x == 0 && flag == 0) keepw[i >> 5] |= (1u << (i & 31));
        __syncthreads();
    }
    for (int t = threadIdx.x; t < W148; t += blockDim.x) keepOut[t] = keepw[t];
}

// ---------------- final top-1000 assembly ----------------
__global__ void k_final(const unsigned* __restrict__ keep,
                        const float* __restrict__ sbox, const float* __restrict__ ssc,
                        float* __restrict__ out) {
    if (blockIdx.x != 0 || threadIdx.x != 0) return;
    int q = 0;
    for (int i = 0; i < NTOT_CAT && q < 1000; ++i) {
        if ((keep[i >> 5] >> (i & 31)) & 1u) {
            out[q * 4 + 0] = sbox[(size_t)i * 4 + 0];
            out[q * 4 + 1] = sbox[(size_t)i * 4 + 1];
            out[q * 4 + 2] = sbox[(size_t)i * 4 + 2];
            out[q * 4 + 3] = sbox[(size_t)i * 4 + 3];
            out[4000 + q] = ssc[i];
            ++q;
        }
    }
    for (int i = 0; i < NTOT_CAT && q < 1000; ++i) {
        if (!((keep[i >> 5] >> (i & 31)) & 1u)) {
            out[q * 4 + 0] = sbox[(size_t)i * 4 + 0];
            out[q * 4 + 1] = sbox[(size_t)i * 4 + 1];
            out[q * 4 + 2] = sbox[(size_t)i * 4 + 2];
            out[q * 4 + 3] = sbox[(size_t)i * 4 + 3];
            out[4000 + q] = -2.0f;
            ++q;
        }
    }
}

// ---------------- host driver ----------------
extern "C" void kernel_launch(void* const* d_in, const int* in_sizes, int n_in,
                              void* d_out, int out_size, void* d_ws, size_t ws_size,
                              hipStream_t stream) {
    (void)in_sizes; (void)n_in; (void)out_size; (void)ws_size;
    const float* x[5];
    for (int l = 0; l < 5; ++l) x[l] = (const float*)d_in[l];
    const float* conv_w  = (const float*)d_in[5];
    const float* conv_b  = (const float*)d_in[6];
    const float* obj_w   = (const float*)d_in[7];
    const float* obj_b   = (const float*)d_in[8];
    const float* delta_w = (const float*)d_in[9];
    const float* delta_b = (const float*)d_in[10];
    float* out = (float*)d_out;

    static const int   Hs[5]   = {192, 96, 48, 24, 12};
    static const int   Ws_[5]  = {320, 160, 80, 40, 20};
    static const int   hw[5]   = {61440, 15360, 3840, 960, 240};
    static const float strd[5] = {4.f, 8.f, 16.f, 32.f, 64.f};
    static const float asz[5]  = {32.f, 64.f, 128.f, 256.f, 512.f};
    static const int   nan_[5] = {184320, 46080, 11520, 2880, 720};
    static const int   kper[5] = {1000, 1000, 1000, 1000, 720};

    int posoff[5], aoff[5], catoff[5];
    int ps = 0, as_ = 0, cs = 0;
    for (int l = 0; l < 5; ++l) {
        posoff[l] = ps; ps  += hw[l];
        aoff[l]   = as_; as_ += nan_[l];
        catoff[l] = cs; cs  += kper[l];
    }
    const int S = ps;       // 81840
    const int A = as_;      // 245520

    char* base = (char*)d_ws;
    size_t off = 0;
    auto alloc = [&](size_t bytes) -> char* {
        char* p = base + off;
        off = (off + bytes + 255) & ~(size_t)255;
        return p;
    };
    _Float16* xhwc  = (_Float16*)alloc((size_t)S * 256 * 2);
    _Float16* wpk   = (_Float16*)alloc((size_t)256 * 2304 * 2);
    _Float16* feat  = (_Float16*)alloc((size_t)S * 256 * 2);
    float* sc_all   = (float*)alloc((size_t)A * 4);
    float* bx_all   = (float*)alloc((size_t)A * 16);
    int* hist       = (int*)alloc((size_t)5 * 4096 * 4);
    int* selT       = (int*)alloc(5 * 4);
    int* selR       = (int*)alloc(5 * 4);
    int* selC       = (int*)alloc(10 * 4);
    float* selsc    = (float*)alloc((size_t)NSORT * 4);
    float* selbox   = (float*)alloc((size_t)NSORT * 16);
    float* keys     = (float*)alloc((size_t)NSORT * 4);
    int* sidx       = (int*)alloc((size_t)NSORT * 4);
    float* sbox     = (float*)alloc((size_t)NTOT_CAT * 16);
    float* ssc      = (float*)alloc((size_t)NTOT_CAT * 4);
    unsigned* sup   = (unsigned*)alloc((size_t)NTOT_CAT * W148 * 4);
    unsigned* keep  = (unsigned*)alloc(W148 * 4);

    // 1) layout conversions
    for (int l = 0; l < 5; ++l) {
        int n = hw[l] * 256;
        k_convert<<<(n + 255) / 256, 256, 0, stream>>>(x[l], xhwc + (size_t)posoff[l] * 256, hw[l]);
    }
    k_packw<<<(256 * 2304 + 255) / 256, 256, 0, stream>>>(conv_w, wpk);

    // 2) WMMA conv + ReLU per level (128x128 tile)
    for (int l = 0; l < 5; ++l) {
        int mtiles = (hw[l] + 127) / 128;
        dim3 grid(mtiles, 2);
        k_conv<<<grid, 256, 0, stream>>>(xhwc + (size_t)posoff[l] * 256, wpk, conv_b,
                                         feat + (size_t)posoff[l] * 256,
                                         Hs[l], Ws_[l], hw[l]);
    }

    // 3) heads: scores + decoded boxes
    for (int l = 0; l < 5; ++l) {
        int n = nan_[l];
        k_heads<<<(n + 127) / 128, 128, 0, stream>>>(feat + (size_t)posoff[l] * 256,
                                                     obj_w, obj_b, delta_w, delta_b,
                                                     sc_all + aoff[l], bx_all + (size_t)aoff[l] * 4,
                                                     hw[l], Ws_[l], strd[l], asz[l], n);
    }

    // 4) per-level top-k selection
    k_histzero<<<(5 * 4096 + 255) / 256, 256, 0, stream>>>(hist);
    for (int l = 0; l < 5; ++l)
        k_hist<<<(nan_[l] + 255) / 256, 256, 0, stream>>>(sc_all + aoff[l], hist + l * 4096, nan_[l]);
    k_thresh<<<5, 32, 0, stream>>>(hist, selT, selR, selC);
    for (int l = 0; l < 5; ++l)
        k_compact<<<(nan_[l] + 255) / 256, 256, 0, stream>>>(sc_all + aoff[l],
                                                             bx_all + (size_t)aoff[l] * 4,
                                                             selsc, selbox, selT, selR, selC,
                                                             l, nan_[l], catoff[l], kper[l]);

    // 5) global descending sort of the 4720 candidates
    k_sortinit<<<NSORT / 256, 256, 0, stream>>>(keys, sidx, selsc);
    for (int k2 = 2; k2 <= NSORT; k2 <<= 1)
        for (int j = k2 >> 1; j >= 1; j >>= 1)
            k_bitonic<<<NSORT / 256, 256, 0, stream>>>(keys, sidx, j, k2);
    k_gather<<<(NTOT_CAT + 255) / 256, 256, 0, stream>>>(keys, sidx, selbox, sbox, ssc);

    // 6) NMS
    int supn = NTOT_CAT * W148;
    k_supmask<<<(supn + 255) / 256, 256, 0, stream>>>(sbox, sup);
    k_nms<<<1, 256, 0, stream>>>(sup, keep);

    // 7) final output assembly
    k_final<<<1, 32, 0, stream>>>(keep, sbox, ssc, out);
}
